// GATChessModel_20495583936864
// MI455X (gfx1250) — matrix operand order repaired and checked by reading, not compiled, not run on hardware
//
#include <hip/hip_runtime.h>
#include <hip/hip_bf16.h>
#include <math.h>

// ---------------------------------------------------------------------------
// 2-layer GAT for MI455X (gfx1250, wave32).
//  - GEMM1 via V_WMMA_F32_16X16X32_BF16 using bf16-hi/lo split (fp32 accuracy)
//  - edge softmax restructured to 2 passes (max, then exp-weighted accumulate)
//  - edge aggregation: 1 wave per edge, 32 lanes = 32 channels, f32 atomics to
//    L2 (xp is 51MB -> L2 resident on the 192MB L2)
// ---------------------------------------------------------------------------

#define F_IN 128
#define H1 8
#define C1 32
#define F_MID 256          // H1*C1
#define NEG_SLOPE 0.2f

typedef __attribute__((ext_vector_type(16))) __bf16 v16bf;
typedef __attribute__((ext_vector_type(8)))  __bf16 v8bf;
typedef __attribute__((ext_vector_type(8)))  float  v8f;

// ---------------- utility ----------------

__global__ void fill_f32(float* __restrict__ p, float v, int n) {
  int i = blockIdx.x * blockDim.x + threadIdx.x;
  if (i < n) p[i] = v;
}

__device__ __forceinline__ void atomicMaxF(float* addr, float val) {
  // monotonic-bit-pattern trick; memory is initialized to -inf
  if (val >= 0.0f) atomicMax((int*)addr, __float_as_int(val));
  else             atomicMin((unsigned int*)addr, (unsigned int)__float_as_int(val));
}

__device__ __forceinline__ float lrelu(float x) {
  return x > 0.0f ? x : NEG_SLOPE * x;
}

// ---------------- bf16 hi/lo split (fp32 -> bf16x2) ----------------

__global__ void split_bf16(const float* __restrict__ x, __bf16* __restrict__ hi,
                           __bf16* __restrict__ lo, int n) {
  int i = blockIdx.x * blockDim.x + threadIdx.x;
  if (i >= n) return;
  float v = x[i];
  __bf16 h = (__bf16)v;
  hi[i] = h;
  lo[i] = (__bf16)(v - (float)h);
}

// W[rows,cols] (row major) -> hiT/loT[cols,rows] (transposed, bf16)
__global__ void split_bf16_T(const float* __restrict__ w, __bf16* __restrict__ hiT,
                             __bf16* __restrict__ loT, int rows, int cols) {
  int i = blockIdx.x * blockDim.x + threadIdx.x;
  if (i >= rows * cols) return;
  int r = i / cols, c = i % cols;
  float v = w[i];
  __bf16 h = (__bf16)v;
  hiT[(size_t)c * rows + r] = h;
  loT[(size_t)c * rows + r] = (__bf16)(v - (float)h);
}

// ---------------- GEMM1: xp[N,256] = x[N,128] @ W1[128,256] via WMMA ----------------
// One wave per 16x16 output tile; K loop 128 in steps of 32; bf16x3 split.
__global__ __launch_bounds__(256) void gemm1_wmma(
    const __bf16* __restrict__ Ahi, const __bf16* __restrict__ Alo,
    const __bf16* __restrict__ BhiT, const __bf16* __restrict__ BloT,
    float* __restrict__ xp, int nrows) {
  const int lane = threadIdx.x & 31;
  const int wave = threadIdx.x >> 5;
  const int tile = blockIdx.x * 8 + wave;
  const int nt   = tile & 15;        // 256/16 = 16 column tiles
  const int mt   = tile >> 4;
  if (mt * 16 >= nrows) return;
  const int half = lane >> 4;        // 0: lanes 0-15, 1: lanes 16-31
  const int l15  = lane & 15;
  const int m    = mt * 16 + l15;    // A row for this lane
  const int col  = nt * 16 + l15;    // B/D column for this lane

  const __bf16* arow_hi = Ahi  + (size_t)m   * F_IN;
  const __bf16* arow_lo = Alo  + (size_t)m   * F_IN;
  const __bf16* bcol_hi = BhiT + (size_t)col * F_IN;
  const __bf16* bcol_lo = BloT + (size_t)col * F_IN;

  v8f c = {};
#pragma unroll
  for (int kk = 0; kk < F_IN; kk += 32) {
    // A fragment (16-bit A 16x32 layout): lane holds K = {kk+8*half .. +8} and
    // {kk+16+8*half .. +8} -> two contiguous 16B loads.
    v16bf ahi, alo, bhi, blo;
    ((v8bf*)&ahi)[0] = *(const v8bf*)(arow_hi + kk + 8 * half);
    ((v8bf*)&ahi)[1] = *(const v8bf*)(arow_hi + kk + 16 + 8 * half);
    ((v8bf*)&alo)[0] = *(const v8bf*)(arow_lo + kk + 8 * half);
    ((v8bf*)&alo)[1] = *(const v8bf*)(arow_lo + kk + 16 + 8 * half);
    // B fragment: lane holds column `col`, K = {kk+8*half .. +8} and
    // {kk+16+8*half .. +8} (same as A), contiguous because W1 was staged transposed.
    ((v8bf*)&bhi)[0] = *(const v8bf*)(bcol_hi + kk + 8 * half);
    ((v8bf*)&bhi)[1] = *(const v8bf*)(bcol_hi + kk + 16 + 8 * half);
    ((v8bf*)&blo)[0] = *(const v8bf*)(bcol_lo + kk + 8 * half);
    ((v8bf*)&blo)[1] = *(const v8bf*)(bcol_lo + kk + 16 + 8 * half);

    c = __builtin_amdgcn_wmma_f32_16x16x32_bf16(false, ahi, false, bhi, (short)0, c, false, false);
    c = __builtin_amdgcn_wmma_f32_16x16x32_bf16(false, ahi, false, blo, (short)0, c, false, false);
    c = __builtin_amdgcn_wmma_f32_16x16x32_bf16(false, alo, false, bhi, (short)0, c, false, false);
  }
  // C/D layout: VGPR r -> row mt*16 + r + 8*half, column col
  float* outp = xp + (size_t)(mt * 16 + half * 8) * F_MID + col;
#pragma unroll
  for (int r = 0; r < 8; ++r) outp[(size_t)r * F_MID] = c[r];
}

// ---------------- per-node attention dots: a_s[n,h], a_d[n,h] ----------------
__global__ void node_attn1(const float* __restrict__ xp, const float* __restrict__ att_s,
                           const float* __restrict__ att_d, float* __restrict__ as,
                           float* __restrict__ ad, int n) {
  int i = blockIdx.x * blockDim.x + threadIdx.x;  // over n*H1
  if (i >= n * H1) return;
  int node = i >> 3, h = i & 7;
  const float* row = xp + (size_t)node * F_MID + h * C1;
  const float* s   = att_s + h * C1;
  const float* d   = att_d + h * C1;
  float accS = 0.0f, accD = 0.0f;
#pragma unroll
  for (int c2 = 0; c2 < C1; ++c2) { float v = row[c2]; accS += v * s[c2]; accD += v * d[c2]; }
  as[i] = accS;
  ad[i] = accD;
}

// ---------------- edge helpers ----------------
__device__ __forceinline__ void edge_sd(const int* __restrict__ src, const int* __restrict__ dst,
                                        int E, int idx, int& s, int& d) {
  if (idx < E) { s = src[idx]; d = dst[idx]; }
  else         { s = idx - E;  d = idx - E; }     // self loop
}

// pass 1: per-dst segment max of leaky_relu(a_s[src]+a_d[dst])
__global__ void edge1_max(const int* __restrict__ ei, int E, int Etot,
                          const float* __restrict__ as, const float* __restrict__ ad,
                          float* __restrict__ m) {
  int idx = blockIdx.x * blockDim.x + threadIdx.x;
  if (idx >= Etot) return;
  int s, d; edge_sd(ei, ei + E, E, idx, s, d);
#pragma unroll
  for (int h = 0; h < H1; ++h) {
    float e = lrelu(as[s * H1 + h] + ad[d * H1 + h]);
    atomicMaxF(&m[d * H1 + h], e);
  }
}

// pass 2: denom[d,h] += ex ; accum[d,h,c] += ex * xp[s,h,c]  (wave per edge)
__global__ __launch_bounds__(256) void edge1_accum(
    const int* __restrict__ ei, int E, int Etot,
    const float* __restrict__ as, const float* __restrict__ ad,
    const float* __restrict__ m, const float* __restrict__ xp,
    float* __restrict__ denom, float* __restrict__ accum) {
  int edge = blockIdx.x * 8 + (threadIdx.x >> 5);
  int lane = threadIdx.x & 31;
  if (edge >= Etot) return;
  int s, d; edge_sd(ei, ei + E, E, edge, s, d);
  const float* xs  = xp    + (size_t)s * F_MID;
  float*       acc = accum + (size_t)d * F_MID;
#pragma unroll
  for (int h = 0; h < H1; ++h) {
    float e  = lrelu(as[s * H1 + h] + ad[d * H1 + h]);
    float ex = __expf(e - m[d * H1 + h]);
    if (lane == 0) atomicAdd(&denom[d * H1 + h], ex);
    atomicAdd(&acc[h * C1 + lane], ex * xs[h * C1 + lane]);
  }
}

// finalize layer 1 (in place): h = ELU(accum/denom + b1)
__global__ void final1(float* __restrict__ accum, const float* __restrict__ denom,
                       const float* __restrict__ b, int n) {
  int i = blockIdx.x * blockDim.x + threadIdx.x;
  if (i >= n * F_MID) return;
  int node = i >> 8;
  int j    = i & 255;
  int h    = j >> 5;
  float v = accum[i] / denom[node * H1 + h] + b[j];
  accum[i] = v > 0.0f ? v : (__expf(v) - 1.0f);   // ELU
}

// ---------------- layer 2 ----------------
// xp2[n] = dot(h[n,:256], W2), a_s2/a_d2 scalars. Wave per node, butterfly reduce.
__global__ __launch_bounds__(256) void gemm2_attn(
    const float* __restrict__ h, const float* __restrict__ W2,
    const float* __restrict__ att_s2, const float* __restrict__ att_d2,
    float* __restrict__ xp2, float* __restrict__ as2, float* __restrict__ ad2, int n) {
  int node = blockIdx.x * 8 + (threadIdx.x >> 5);
  int lane = threadIdx.x & 31;
  if (node >= n) return;
  const float* row = h + (size_t)node * F_MID;
  float acc = 0.0f;
#pragma unroll
  for (int j = 0; j < F_MID; j += 32) acc += row[j + lane] * W2[j + lane];
#pragma unroll
  for (int off = 16; off > 0; off >>= 1) acc += __shfl_xor(acc, off, 32);
  if (lane == 0) {
    xp2[node] = acc;
    as2[node] = acc * att_s2[0];
    ad2[node] = acc * att_d2[0];
  }
}

__global__ void edge2_max(const int* __restrict__ ei, int E, int Etot,
                          const float* __restrict__ as2, const float* __restrict__ ad2,
                          float* __restrict__ m2) {
  int idx = blockIdx.x * blockDim.x + threadIdx.x;
  if (idx >= Etot) return;
  int s, d; edge_sd(ei, ei + E, E, idx, s, d);
  atomicMaxF(&m2[d], lrelu(as2[s] + ad2[d]));
}

__global__ void edge2_accum(const int* __restrict__ ei, int E, int Etot,
                            const float* __restrict__ as2, const float* __restrict__ ad2,
                            const float* __restrict__ m2, const float* __restrict__ xp2,
                            float* __restrict__ denom2, float* __restrict__ accum2) {
  int idx = blockIdx.x * blockDim.x + threadIdx.x;
  if (idx >= Etot) return;
  int s, d; edge_sd(ei, ei + E, E, idx, s, d);
  float e  = lrelu(as2[s] + ad2[d]);
  float ex = __expf(e - m2[d]);
  atomicAdd(&denom2[d], ex);
  atomicAdd(&accum2[d], ex * xp2[s]);
}

__global__ void final2(const float* __restrict__ accum2, const float* __restrict__ denom2,
                       const float* __restrict__ b2, float* __restrict__ out, int n) {
  int i = blockIdx.x * blockDim.x + threadIdx.x;
  if (i >= n) return;
  out[i] = accum2[i] / denom2[i] + b2[0];
}

// ---------------- launcher ----------------

extern "C" void kernel_launch(void* const* d_in, const int* in_sizes, int n_in,
                              void* d_out, int out_size, void* d_ws, size_t ws_size,
                              hipStream_t stream) {
  const float* x        = (const float*)d_in[0];
  const int*   ei       = (const int*)  d_in[1];   // [2,E]: src then dst
  const float* W1       = (const float*)d_in[2];   // [128,256]
  const float* att_s1   = (const float*)d_in[3];   // [8,32]
  const float* att_d1   = (const float*)d_in[4];
  const float* b1       = (const float*)d_in[5];   // [256]
  const float* W2       = (const float*)d_in[6];   // [256,1]
  const float* att_s2   = (const float*)d_in[7];   // [1,1]
  const float* att_d2   = (const float*)d_in[8];
  const float* b2       = (const float*)d_in[9];
  float*       out      = (float*)d_out;

  const int N    = in_sizes[0] / F_IN;
  const int E    = in_sizes[1] / 2;
  const int Etot = E + N;                 // + self loops

  // carve workspace
  size_t off = 0;
  auto carve = [&](size_t bytes) -> void* {
    void* p = (char*)d_ws + off;
    off += (bytes + 255) & ~(size_t)255;
    return p;
  };
  __bf16* x_hi   = (__bf16*)carve((size_t)N * F_IN * 2);
  __bf16* x_lo   = (__bf16*)carve((size_t)N * F_IN * 2);
  __bf16* w1_hiT = (__bf16*)carve((size_t)F_IN * F_MID * 2);
  __bf16* w1_loT = (__bf16*)carve((size_t)F_IN * F_MID * 2);
  float*  xp1    = (float*) carve((size_t)N * F_MID * 4);
  float*  as1    = (float*) carve((size_t)N * H1 * 4);
  float*  ad1    = (float*) carve((size_t)N * H1 * 4);
  float*  m1     = (float*) carve((size_t)N * H1 * 4);
  float*  den1   = (float*) carve((size_t)N * H1 * 4);
  float*  acc1   = (float*) carve((size_t)N * F_MID * 4);   // becomes h in place
  float*  xp2    = (float*) carve((size_t)N * 4);
  float*  as2    = (float*) carve((size_t)N * 4);
  float*  ad2    = (float*) carve((size_t)N * 4);
  float*  m2     = (float*) carve((size_t)N * 4);
  float*  den2   = (float*) carve((size_t)N * 4);
  float*  acc2   = (float*) carve((size_t)N * 4);
  (void)ws_size;

  const int T = 256;
  auto blocks = [](int n, int t) { return (n + t - 1) / t; };

  // init accumulators (every call -> deterministic)
  fill_f32<<<blocks(N * H1, T),    T, 0, stream>>>(m1,  -INFINITY, N * H1);
  fill_f32<<<blocks(N * H1, T),    T, 0, stream>>>(den1, 0.0f,     N * H1);
  fill_f32<<<blocks(N * F_MID, T), T, 0, stream>>>(acc1, 0.0f,     N * F_MID);
  fill_f32<<<blocks(N, T),         T, 0, stream>>>(m2,  -INFINITY, N);
  fill_f32<<<blocks(N, T),         T, 0, stream>>>(den2, 0.0f,     N);
  fill_f32<<<blocks(N, T),         T, 0, stream>>>(acc2, 0.0f,     N);

  // bf16 split staging
  split_bf16  <<<blocks(N * F_IN, T), T, 0, stream>>>(x, x_hi, x_lo, N * F_IN);
  split_bf16_T<<<blocks(F_IN * F_MID, T), T, 0, stream>>>(W1, w1_hiT, w1_loT, F_IN, F_MID);

  // layer 1
  const int tilesM = (N + 15) / 16;            // 3125 (exact)
  const int tiles  = tilesM * (F_MID / 16);    // 50000 waves
  gemm1_wmma<<<blocks(tiles, 8), T, 0, stream>>>(x_hi, x_lo, w1_hiT, w1_loT, xp1, N);
  node_attn1<<<blocks(N * H1, T), T, 0, stream>>>(xp1, att_s1, att_d1, as1, ad1, N);
  edge1_max  <<<blocks(Etot, T), T, 0, stream>>>(ei, E, Etot, as1, ad1, m1);
  edge1_accum<<<blocks(Etot, 8), T, 0, stream>>>(ei, E, Etot, as1, ad1, m1, xp1, den1, acc1);
  final1<<<blocks(N * F_MID, T), T, 0, stream>>>(acc1, den1, b1, N);   // acc1 := h

  // layer 2
  gemm2_attn <<<blocks(N, 8), T, 0, stream>>>(acc1, W2, att_s2, att_d2, xp2, as2, ad2, N);
  edge2_max  <<<blocks(Etot, T), T, 0, stream>>>(ei, E, Etot, as2, ad2, m2);
  edge2_accum<<<blocks(Etot, T), T, 0, stream>>>(ei, E, Etot, as2, ad2, m2, xp2, den2, acc2);
  final2<<<blocks(N, T), T, 0, stream>>>(acc2, den2, b2, out, N);
}
